// Network_81862076662591
// MI455X (gfx1250) — matrix operand order repaired and checked by reading, not compiled, not run on hardware
//
#include <hip/hip_runtime.h>
#include <hip/hip_bf16.h>
#include <math.h>

typedef __attribute__((ext_vector_type(16))) _Float16 v16h;
typedef __attribute__((ext_vector_type(8)))  _Float16 v8h;
typedef __attribute__((ext_vector_type(8)))  float    v8f;

#define NWAVES 16
#define BLK    512
#define NBLOCKS 16
#define NEGINF -1e9f

__device__ __forceinline__ float sigmoidf(float x) { return 1.f / (1.f + expf(-x)); }

// ---------------- fragment helpers ----------------

// A fragment: 16x32 f16 from row-major LDS. Lane l: row = l&15,
// K chunk [kb..kb+15], kb = k0 + (l>=16 ? 16 : 0).
__device__ __forceinline__ v16h load_A_frag(const _Float16* A, int lda, int k0, int lane) {
  const int row = lane & 15;
  const int kb  = k0 + ((lane >> 4) << 4);
  const _Float16* p = A + row * lda + kb;
  v8h lo = *(const v8h*)(p);
  v8h hi = *(const v8h*)(p + 8);
  v16h a;
#pragma unroll
  for (int j = 0; j < 8; ++j) { a[j] = lo[j]; a[8 + j] = hi[j]; }
  return a;
}

// Direct (fallback) B fragment from fp32 row-major W[K x N].
__device__ __forceinline__ v16h load_B_direct(const float* W, int K, int N, int nb, int k0, int lane) {
  const int col = nb + (lane & 15);
  const int kb  = k0 + ((lane >> 4) << 4);
  v16h b;
#pragma unroll
  for (int j = 0; j < 16; ++j) {
    const int k = kb + j;
    float w = (k < K && col < N) ? W[(size_t)k * N + col] : 0.f;
    b[j] = (_Float16)w;
  }
  return b;
}

__device__ __forceinline__ v8f init_bias(const float* bias, int N, int nb, int lane) {
  const int col = nb + (lane & 15);
  const float bv = (col < N) ? bias[col] : 0.f;
  v8f acc = {bv, bv, bv, bv, bv, bv, bv, bv};
  return acc;
}

// acc += A[16xK] * W[KxN] tile. If Bp != nullptr, B fragments come pre-packed
// (f16, WMMA lane layout): fragment f = (nb/16)*fragsK + ks at Bp + (f*32+lane)*16.
__device__ __forceinline__ v8f gemm_acc(v8f acc, const _Float16* A, int lda,
                                        const float* W, const _Float16* Bp,
                                        int K, int Kp, int N, int nb, int lane) {
  const int fragsK = Kp >> 5;
  if (Bp) {
    const _Float16* bb = Bp + (((size_t)(nb >> 4) * fragsK) * 32 + lane) * 16;
    for (int ks = 0; ks < fragsK; ++ks) {
      v16h a = load_A_frag(A, lda, ks << 5, lane);
      v8h blo = *(const v8h*)(bb);
      v8h bhi = *(const v8h*)(bb + 8);
      v16h b;
#pragma unroll
      for (int j = 0; j < 8; ++j) { b[j] = blo[j]; b[8 + j] = bhi[j]; }
      acc = __builtin_amdgcn_wmma_f32_16x16x32_f16(false, a, false, b, (short)0, acc, false, false);
      bb += 512;
    }
  } else {
    for (int k0 = 0; k0 < Kp; k0 += 32) {
      v16h a = load_A_frag(A, lda, k0, lane);
      v16h b = load_B_direct(W, K, N, nb, k0, lane);
      acc = __builtin_amdgcn_wmma_f32_16x16x32_f16(false, a, false, b, (short)0, acc, false, false);
    }
  }
  return acc;
}

// C/D layout: VGPR e -> row = e + 8*(lane>=16), col = lane&15.
__device__ __forceinline__ void store_tile_f16(v8f acc, _Float16* O, int ldo, int nb, int lane, bool leaky) {
  const int col = nb + (lane & 15);
  const int rb  = (lane >> 4) << 3;
#pragma unroll
  for (int e = 0; e < 8; ++e) {
    float v = acc[e];
    if (leaky) v = v > 0.f ? v : 0.2f * v;
    O[(size_t)(rb + e) * ldo + col] = (_Float16)v;
  }
}

__device__ __forceinline__ void store_tile_f32(v8f acc, float* O, int ldo, int nb, int lane) {
  const int col = nb + (lane & 15);
  const int rb  = (lane >> 4) << 3;
#pragma unroll
  for (int e = 0; e < 8; ++e) O[(size_t)(rb + e) * ldo + col] = acc[e];
}

// Prefetch a contiguous region into L2 (global_prefetch_b8); each caller thread
// takes disjoint 128B lines; `gt` is the grid-global thread id, `gn` grid size.
__device__ __forceinline__ void prefetch_region(const _Float16* p, size_t elems,
                                                size_t gt, size_t gn) {
  const char* c = (const char*)p;
  const size_t bytes = elems * sizeof(_Float16);
  for (size_t off = gt * 128; off < bytes; off += gn * 128)
    __builtin_prefetch(c + off, 0, 1);
}

// ---------------- weight pre-pack kernel ----------------
// One thread per (t, nt, ks, lane): writes 16 f16 (32B contiguous) in WMMA B layout.
__global__ void pack_weights(const float* __restrict__ src, _Float16* __restrict__ dst,
                             int T, int K, int N, int Kp, int Np) {
  const int fragsK = Kp >> 5;
  const int fragsN = Np >> 4;
  const size_t lanesPerT = (size_t)fragsN * fragsK * 32;
  const size_t total = (size_t)T * lanesPerT;
  size_t idx = (size_t)blockIdx.x * blockDim.x + threadIdx.x;
  if (idx >= total) return;
  const int    t    = (int)(idx / lanesPerT);
  const size_t r    = idx % lanesPerT;
  const int    lane = (int)(r & 31);
  const size_t f    = r >> 5;              // nt*fragsK + ks
  const int    ks   = (int)(f % fragsK);
  const int    nt   = (int)(f / fragsK);
  const int col = nt * 16 + (lane & 15);
  const int kb  = (ks << 5) + ((lane >> 4) << 4);
  const float* W = src + (size_t)t * K * N;
  _Float16* o = dst + (size_t)t * Np * Kp + (f * 32 + lane) * 16;
  v8h lo, hi;
#pragma unroll
  for (int j = 0; j < 8; ++j) {
    const int k0 = kb + j, k1 = kb + 8 + j;
    lo[j] = (_Float16)((k0 < K && col < N) ? W[(size_t)k0 * N + col] : 0.f);
    hi[j] = (_Float16)((k1 < K && col < N) ? W[(size_t)k1 * N + col] : 0.f);
  }
  *(v8h*)(o)     = lo;
  *(v8h*)(o + 8) = hi;
}

// ---------------- main persistent scan kernel ----------------
__global__ __launch_bounds__(BLK, 1) void dps_scan_kernel(
    const float* __restrict__ x,   const float* __restrict__ gum,
    const float* __restrict__ W1,  const float* __restrict__ b1,
    const float* __restrict__ W2,  const float* __restrict__ b2,
    const float* __restrict__ W3,  const float* __restrict__ b3,
    const float* __restrict__ W4,  const float* __restrict__ b4,
    const float* __restrict__ Wf2, const float* __restrict__ bf2,
    const float* __restrict__ Wg1, const float* __restrict__ bg1,
    const float* __restrict__ Wg2, const float* __restrict__ bg2,
    const float* __restrict__ Wih, const float* __restrict__ Whh,
    const float* __restrict__ bih, const float* __restrict__ bhh,
    const _Float16* __restrict__ pW1, const _Float16* __restrict__ pW2,
    const _Float16* __restrict__ pW3, const _Float16* __restrict__ pW4,
    const _Float16* __restrict__ pWf2, const _Float16* __restrict__ pWg1,
    const _Float16* __restrict__ pWg2, const _Float16* __restrict__ pWih,
    const _Float16* __restrict__ pWhh,
    float* __restrict__ out)
{
  __shared__ __align__(16) float         sScr[16][784];  // gates(512)/logits(784) fp32
  __shared__ __align__(16) _Float16      sX16[16][784];  // x rows f16
  __shared__ __align__(16) _Float16      sP0[16][800];   // g1 / a1 (tail zero)
  __shared__ __align__(16) _Float16      sP1[16][800];   // y  / a2 (tail zero)
  __shared__ __align__(16) _Float16      sH[16][128];
  __shared__ __align__(16) _Float16      sLin[16][128];
  __shared__ __align__(16) _Float16      sT0[16][128];
  __shared__ __align__(16) float         sC[16][128];
  __shared__ unsigned char               sMem[16][784];

  const int tid  = threadIdx.x;
  const int lane = tid & 31;
  const int wid  = tid >> 5;
  const int r0   = blockIdx.x * 16;
  const size_t gt = (size_t)blockIdx.x * BLK + tid;   // grid-global thread id
  const size_t gn = (size_t)NBLOCKS * BLK;

  for (int i = tid; i < 16 * 784; i += BLK) {
    int r = i / 784, d = i % 784;
    sX16[r][d] = (_Float16)x[(size_t)(r0 + r) * 784 + d];
    sMem[r][d] = 0;
  }
  for (int i = tid; i < 16 * 800; i += BLK) {
    int r = i / 800, d = i % 800;
    sP0[r][d] = (_Float16)0.f; sP1[r][d] = (_Float16)0.f;
  }
  for (int i = tid; i < 16 * 128; i += BLK) {
    int r = i >> 7, j = i & 127;
    sH[r][j] = (_Float16)0.f; sLin[r][j] = (_Float16)0.f; sC[r][j] = 0.f;
  }
  __syncthreads();

  for (int t = 0; t < 100; ++t) {
    const float* W1t  = W1  + (size_t)t * 784 * 784;  const float* b1t  = b1  + (size_t)t * 784;
    const float* W2t  = W2  + (size_t)t * 784 * 256;  const float* b2t  = b2  + (size_t)t * 256;
    const float* W3t  = W3  + (size_t)t * 256 * 128;  const float* b3t  = b3  + (size_t)t * 128;
    const float* W4t  = W4  + (size_t)t * 128 * 128;  const float* b4t  = b4  + (size_t)t * 128;
    const float* Wf2t = Wf2 + (size_t)t * 128 * 10;   const float* bf2t = bf2 + (size_t)t * 10;
    const float* Wg1t = Wg1 + (size_t)t * 128 * 256;  const float* bg1t = bg1 + (size_t)t * 256;
    const float* Wg2t = Wg2 + (size_t)t * 256 * 784;  const float* bg2t = bg2 + (size_t)t * 784;
    const _Float16* pW1t  = pW1  ? pW1  + (size_t)t * 800 * 784 : nullptr;
    const _Float16* pW2t  = pW2  ? pW2  + (size_t)t * 800 * 256 : nullptr;
    const _Float16* pW3t  = pW3  ? pW3  + (size_t)t * 256 * 128 : nullptr;
    const _Float16* pW4t  = pW4  ? pW4  + (size_t)t * 128 * 128 : nullptr;
    const _Float16* pWf2t = pWf2 ? pWf2 + (size_t)t * 128 * 16  : nullptr;
    const _Float16* pWg1t = pWg1 ? pWg1 + (size_t)t * 128 * 256 : nullptr;
    const _Float16* pWg2t = pWg2 ? pWg2 + (size_t)t * 256 * 784 : nullptr;

    // L2 prefetch of NEXT step's packed weights (global_prefetch_b8):
    // all blocks/threads cover disjoint 128B lines; hidden behind this step's GEMMs.
    if (pW1 && t + 1 < 100) {
      const size_t n1 = (size_t)(t + 1);
      prefetch_region(pW1  + n1 * 800 * 784, (size_t)800 * 784, gt, gn);
      prefetch_region(pW2  + n1 * 800 * 256, (size_t)800 * 256, gt, gn);
      prefetch_region(pWg2 + n1 * 256 * 784, (size_t)256 * 784, gt, gn);
      prefetch_region(pWg1 + n1 * 128 * 256, (size_t)128 * 256, gt, gn);
      prefetch_region(pW3  + n1 * 256 * 128, (size_t)256 * 128, gt, gn);
      prefetch_region(pW4  + n1 * 128 * 128, (size_t)128 * 128, gt, gn);
      prefetch_region(pWf2 + n1 * 128 * 16,  (size_t)128 * 16,  gt, gn);
    }

    // 1) LSTM gates: lin@Wih + h@Whh + bih + bhh  (K=128, N=512)
    for (int tile = wid; tile < 32; tile += NWAVES) {
      const int nb  = tile * 16;
      const int col = nb + (lane & 15);
      const float bv = bih[col] + bhh[col];
      v8f acc = {bv, bv, bv, bv, bv, bv, bv, bv};
      acc = gemm_acc(acc, &sLin[0][0], 128, Wih, pWih, 128, 128, 512, nb, lane);
      acc = gemm_acc(acc, &sH[0][0],   128, Whh, pWhh, 128, 128, 512, nb, lane);
      store_tile_f32(acc, &sScr[0][0], 784, nb, lane);
    }
    __syncthreads();

    // 2) LSTM elementwise (gate order i,f,g,o)
    for (int i = tid; i < 16 * 128; i += BLK) {
      int r = i >> 7, j = i & 127;
      float ig = sScr[r][j], fg = sScr[r][128 + j], gg = sScr[r][256 + j], og = sScr[r][384 + j];
      float c = sigmoidf(fg) * sC[r][j] + sigmoidf(ig) * tanhf(gg);
      float h = sigmoidf(og) * tanhf(c);
      sC[r][j] = c;
      sH[r][j] = (_Float16)h;
    }
    __syncthreads();

    // 3) g1 = leaky(h@Wg1 + bg1): K=128, N=256 -> P0
    for (int tile = wid; tile < 16; tile += NWAVES) {
      v8f acc = init_bias(bg1t, 256, tile * 16, lane);
      acc = gemm_acc(acc, &sH[0][0], 128, Wg1t, pWg1t, 128, 128, 256, tile * 16, lane);
      store_tile_f16(acc, &sP0[0][0], 800, tile * 16, lane, true);
    }
    __syncthreads();

    // 4) logits = g1@Wg2 + bg2: K=256, N=784 -> sScr
    for (int tile = wid; tile < 49; tile += NWAVES) {
      v8f acc = init_bias(bg2t, 784, tile * 16, lane);
      acc = gemm_acc(acc, &sP0[0][0], 800, Wg2t, pWg2t, 256, 256, 784, tile * 16, lane);
      store_tile_f32(acc, &sScr[0][0], 784, tile * 16, lane);
    }
    __syncthreads();

    // 5) Gumbel argmax per row (straight-through soft term cancels exactly)
    {
      const int r = wid;
      const float* grow = gum + ((size_t)t * 256 + r0 + r) * 784;
      float bv = -3.4e38f; int bi = 0;
      for (int d = lane; d < 784; d += 32) {
        float v = sMem[r][d] ? NEGINF : (sScr[r][d] + grow[d]);
        if (v > bv || (v == bv && d < bi)) { bv = v; bi = d; }
      }
#pragma unroll
      for (int off = 16; off; off >>= 1) {
        float ov = __shfl_xor(bv, off, 32);
        int   oi = __shfl_xor(bi, off, 32);
        if (ov > bv || (ov == bv && oi < bi)) { bv = ov; bi = oi; }
      }
      if (lane == 0) sMem[r][bi] = 1;
    }
    __syncthreads();

    // y = new_mem * x -> P1
    for (int i = tid; i < 16 * 784; i += BLK) {
      int r = i / 784, d = i % 784;
      sP1[r][d] = sMem[r][d] ? sX16[r][d] : (_Float16)0.f;
    }
    __syncthreads();

    // 6) a1 = leaky(y@W1 + b1): K=784(800), N=784 -> P0
    for (int tile = wid; tile < 49; tile += NWAVES) {
      v8f acc = init_bias(b1t, 784, tile * 16, lane);
      acc = gemm_acc(acc, &sP1[0][0], 800, W1t, pW1t, 784, 800, 784, tile * 16, lane);
      store_tile_f16(acc, &sP0[0][0], 800, tile * 16, lane, true);
    }
    __syncthreads();

    // 7) a2 = leaky(a1@W2 + b2): K=784(800), N=256 -> P1
    for (int tile = wid; tile < 16; tile += NWAVES) {
      v8f acc = init_bias(b2t, 256, tile * 16, lane);
      acc = gemm_acc(acc, &sP0[0][0], 800, W2t, pW2t, 784, 800, 256, tile * 16, lane);
      store_tile_f16(acc, &sP1[0][0], 800, tile * 16, lane, true);
    }
    __syncthreads();

    // 8) a3 = leaky(a2@W3 + b3): K=256, N=128 -> sT0
    for (int tile = wid; tile < 8; tile += NWAVES) {
      v8f acc = init_bias(b3t, 128, tile * 16, lane);
      acc = gemm_acc(acc, &sP1[0][0], 800, W3t, pW3t, 256, 256, 128, tile * 16, lane);
      store_tile_f16(acc, &sT0[0][0], 128, tile * 16, lane, true);
    }
    __syncthreads();

    // 9) a4 = leaky(a3@W4 + b4): K=128, N=128 -> sLin
    for (int tile = wid; tile < 8; tile += NWAVES) {
      v8f acc = init_bias(b4t, 128, tile * 16, lane);
      acc = gemm_acc(acc, &sT0[0][0], 128, W4t, pW4t, 128, 128, 128, tile * 16, lane);
      store_tile_f16(acc, &sLin[0][0], 128, tile * 16, lane, true);
    }
    __syncthreads();

    // 10) s = a4@Wf2 + bf2: K=128, N=10 -> out[B,10,T]
    if (wid == 0) {
      v8f acc = init_bias(bf2t, 10, 0, lane);
      acc = gemm_acc(acc, &sLin[0][0], 128, Wf2t, pWf2t, 128, 128, 10, 0, lane);
      const int col = lane & 15;
      const int rb  = (lane >> 4) << 3;
      if (col < 10) {
#pragma unroll
        for (int e = 0; e < 8; ++e)
          out[((size_t)(r0 + rb + e) * 10 + col) * 100 + t] = acc[e];
      }
    }
    __syncthreads();
  }
}

// ---------------- launcher ----------------
extern "C" void kernel_launch(void* const* d_in, const int* in_sizes, int n_in,
                              void* d_out, int out_size, void* d_ws, size_t ws_size,
                              hipStream_t stream) {
  (void)in_sizes; (void)n_in; (void)out_size;
  const float* x    = (const float*)d_in[0];
  const float* gum  = (const float*)d_in[1];
  const float* W1   = (const float*)d_in[2];
  const float* b1   = (const float*)d_in[3];
  const float* W2   = (const float*)d_in[4];
  const float* b2   = (const float*)d_in[5];
  const float* W3   = (const float*)d_in[6];
  const float* b3   = (const float*)d_in[7];
  const float* W4   = (const float*)d_in[8];
  const float* b4   = (const float*)d_in[9];
  const float* Wf2  = (const float*)d_in[10];
  const float* bf2  = (const float*)d_in[11];
  const float* Wg1  = (const float*)d_in[12];
  const float* bg1  = (const float*)d_in[13];
  const float* Wg2  = (const float*)d_in[14];
  const float* bg2  = (const float*)d_in[15];
  const float* Wih  = (const float*)d_in[16];
  const float* Whh  = (const float*)d_in[17];
  const float* bih  = (const float*)d_in[18];
  const float* bhh  = (const float*)d_in[19];
  float* out = (float*)d_out;

  // packed-weight workspace layout (f16 elements)
  size_t o = 0;
  const size_t oW1  = o; o += (size_t)100 * 800 * 784;
  const size_t oW2  = o; o += (size_t)100 * 800 * 256;
  const size_t oW3  = o; o += (size_t)100 * 256 * 128;
  const size_t oW4  = o; o += (size_t)100 * 128 * 128;
  const size_t oWf2 = o; o += (size_t)100 * 128 * 16;
  const size_t oWg1 = o; o += (size_t)100 * 128 * 256;
  const size_t oWg2 = o; o += (size_t)100 * 256 * 784;
  const size_t oWih = o; o += (size_t)128 * 512;
  const size_t oWhh = o; o += (size_t)128 * 512;
  const size_t reqBytes = o * sizeof(_Float16);

  const _Float16 *pW1 = nullptr, *pW2 = nullptr, *pW3 = nullptr, *pW4 = nullptr,
                 *pWf2 = nullptr, *pWg1 = nullptr, *pWg2 = nullptr,
                 *pWih = nullptr, *pWhh = nullptr;

  if (d_ws && ws_size >= reqBytes) {
    _Float16* ws = (_Float16*)d_ws;
    struct Job { const float* src; size_t off; int T, K, N, Kp, Np; };
    const Job jobs[9] = {
      { W1,  oW1,  100, 784, 784, 800, 784 },
      { W2,  oW2,  100, 784, 256, 800, 256 },
      { W3,  oW3,  100, 256, 128, 256, 128 },
      { W4,  oW4,  100, 128, 128, 128, 128 },
      { Wf2, oWf2, 100, 128,  10, 128,  16 },
      { Wg1, oWg1, 100, 128, 256, 128, 256 },
      { Wg2, oWg2, 100, 256, 784, 256, 784 },
      { Wih, oWih,   1, 128, 512, 128, 512 },
      { Whh, oWhh,   1, 128, 512, 128, 512 },
    };
    for (int i = 0; i < 9; ++i) {
      const Job& j = jobs[i];
      const size_t threads = (size_t)j.T * (j.Np >> 4) * (j.Kp >> 5) * 32;
      const unsigned grid = (unsigned)((threads + 255) / 256);
      pack_weights<<<grid, 256, 0, stream>>>(j.src, ws + j.off, j.T, j.K, j.N, j.Kp, j.Np);
    }
    pW1  = ws + oW1;  pW2  = ws + oW2;  pW3 = ws + oW3; pW4 = ws + oW4;
    pWf2 = ws + oWf2; pWg1 = ws + oWg1; pWg2 = ws + oWg2;
    pWih = ws + oWih; pWhh = ws + oWhh;
  }

  dps_scan_kernel<<<dim3(NBLOCKS), dim3(BLK), 0, stream>>>(
      x, gum, W1, b1, W2, b2, W3, b3, W4, b4, Wf2, bf2,
      Wg1, bg1, Wg2, bg2, Wih, Whh, bih, bhh,
      pW1, pW2, pW3, pW4, pWf2, pWg1, pWg2, pWih, pWhh, out);
}
